// MoE_60146722013497
// MI455X (gfx1250) — matrix operand order repaired and checked by reading, not compiled, not run on hardware
//
#include <hip/hip_runtime.h>
#include <hip/hip_bf16.h>

#define D_MODEL   512
#define N_EXPERTS 8
#define ROWS_WG   64
#define THREADS   512
#define KSTEP     32
#define NKSTEPS   (D_MODEL / KSTEP)   // 16
#define NTN       4                   // N-tiles per wave
#define XROW      552                 // padded LDS row stride (bf16) for Xs/Hs
#define WPSTRIDE  20                  // padded per-col dwords in Wp
#define WPBUFDW   (D_MODEL * WPSTRIDE)        // dwords per Wp buffer (10240)
#define NWPBUF    3                   // triple-buffered panels (2-deep prefetch)
#define PANELDW   (D_MODEL * (KSTEP / 2))     // dwords per packed global panel (8192)
#define PANELBYTES ((size_t)PANELDW * 4)
#define PACKDW    (2 * N_EXPERTS * NKSTEPS * PANELDW)       // 2M dwords packed weights
#define PACK_TASKS (2 * N_EXPERTS * NKSTEPS * 4 * D_MODEL)  // 524288 uint4 tasks

typedef __bf16 bf16_t;
typedef __attribute__((ext_vector_type(16))) __bf16        v16bf;
typedef __attribute__((ext_vector_type(8)))  float         v8f;
typedef __attribute__((ext_vector_type(4)))  unsigned int  v4u;

__device__ __forceinline__ unsigned short f32_to_bf16_bits(float f) {
    unsigned int u = __builtin_bit_cast(unsigned int, f);
    u += 0x7fffu + ((u >> 16) & 1u);
    return (unsigned short)(u >> 16);
}
__device__ __forceinline__ bf16_t f32_to_bf16(float f) {
    return __builtin_bit_cast(bf16_t, f32_to_bf16_bits(f));
}
__device__ __forceinline__ unsigned int pack_bf16x2(float lo, float hi) {
    return (unsigned int)f32_to_bf16_bits(lo) |
           ((unsigned int)f32_to_bf16_bits(hi) << 16);
}
__device__ __forceinline__ v8f v8f_zero() {
    v8f z = {0.f, 0.f, 0.f, 0.f, 0.f, 0.f, 0.f, 0.f};
    return z;
}

// ---- Prep: pack W1/W2 f32 -> bf16 pair-packed fragment-major panels; WgT ----
__global__ __launch_bounds__(THREADS)
void pack_weights(const float* __restrict__ W1, const float* __restrict__ W2,
                  const float* __restrict__ Wg,
                  unsigned int* __restrict__ wp, float* __restrict__ wgt) {
    int idx = blockIdx.x * blockDim.x + threadIdx.x;
    if (idx < PACK_TASKS) {
        int c  = idx & (D_MODEL - 1);
        int jb = (idx >> 9) & 3;
        int kk = (idx >> 11) & 15;
        int e  = (idx >> 15) & 7;
        int g  = idx >> 18;
        const float* Wsrc = (g == 0 ? W1 : W2) + (size_t)e * D_MODEL * D_MODEL;
        v4u outv;
#pragma unroll
        for (int jj = 0; jj < 4; ++jj) {
            int j = jb * 4 + jj;
            int r = kk * KSTEP + 2 * j;
            outv[jj] = pack_bf16x2(Wsrc[(size_t)r * D_MODEL + c],
                                   Wsrc[(size_t)(r + 1) * D_MODEL + c]);
        }
        size_t dst = ((((size_t)(g * N_EXPERTS + e) * NKSTEPS + kk) * D_MODEL + c) * 16) + jb * 4;
        *(v4u*)(wp + dst) = outv;
    } else {
        int t = idx - PACK_TASKS;
        if (t < N_EXPERTS * D_MODEL) {
            int e = t >> 9;
            int d = t & (D_MODEL - 1);
            wgt[e * D_MODEL + d] = Wg[d * N_EXPERTS + e];
        }
    }
}

// ---- Async stage one 32KB packed panel into (padded) LDS -------------------
__device__ __forceinline__ void async_stage_panel(unsigned long long gbase,
                                                  unsigned int* WpBuf, int tid) {
#pragma unroll
    for (int it = 0; it < 4; ++it) {
        int q4 = tid + it * THREADS;               // 0..2047 16B chunks
        int c = q4 >> 2, jb = q4 & 3;
        unsigned int ldsa = (unsigned int)(size_t)WpBuf +
                            (unsigned int)(c * (WPSTRIDE * 4) + jb * 16);
        unsigned int goff = (unsigned int)(c * 64 + jb * 16);
        asm volatile("global_load_async_to_lds_b128 %0, %1, %2"
                     :: "v"(ldsa), "v"(goff), "s"(gbase) : "memory");
    }
}

// ---- One GEMM pass; triple-buffered async panels, 2-deep prefetch ----------
// Panel stream is continuous across passes: buffer index = (pb3 + kk) % 3.
__device__ __forceinline__ void gemm_pass(const bf16_t* __restrict__ Asrc,
                                          unsigned long long gW,
                                          unsigned long long gWnext,
                                          unsigned int* __restrict__ Wp,
                                          v8f acc[2][NTN],
                                          int tid, int mtp, int q8, int half, int ln,
                                          int pb3) {
    for (int kk = 0; kk < NKSTEPS; ++kk) {
        unsigned int* buf = Wp + ((pb3 + kk) % 3) * WPBUFDW;
        // Async completes in order: <=4 outstanding => panel kk fully resident,
        // panel kk+1 still streaming.
        asm volatile("s_wait_asynccnt 0x4" ::: "memory");
        __syncthreads();
        {   // issue panel kk+2 of the continuous stream
            int np = kk + 2;
            unsigned long long nb = (np < NKSTEPS)
                ? gW + (size_t)np * PANELBYTES
                : gWnext + (size_t)(np - NKSTEPS) * PANELBYTES;
            async_stage_panel(nb, Wp + ((pb3 + np) % 3) * WPBUFDW, tid);
        }

        v16bf afrag[2];
#pragma unroll
        for (int mi = 0; mi < 2; ++mi) {
            const bf16_t* abase = Asrc + (size_t)(mtp * 32 + mi * 16 + ln) * XROW
                                 + kk * KSTEP + half * 8;
            v4u a0 = *(const v4u*)(abase);
            v4u a1 = *(const v4u*)(abase + 16);
            afrag[mi] = __builtin_bit_cast(
                v16bf, __builtin_shufflevector(a0, a1, 0, 1, 2, 3, 4, 5, 6, 7));
        }
#pragma unroll
        for (int nt = 0; nt < NTN; ++nt) {
            int col = q8 * 64 + nt * 16 + ln;
            const unsigned int* bp = buf + col * WPSTRIDE + half * 4;
            v4u b0 = *(const v4u*)(bp);
            v4u b1 = *(const v4u*)(bp + 8);
            v16bf bfrag = __builtin_bit_cast(
                v16bf, __builtin_shufflevector(b0, b1, 0, 1, 2, 3, 4, 5, 6, 7));
            acc[0][nt] = __builtin_amdgcn_wmma_f32_16x16x32_bf16(
                false, afrag[0], false, bfrag, (short)0, acc[0][nt], false, false);
            acc[1][nt] = __builtin_amdgcn_wmma_f32_16x16x32_bf16(
                false, afrag[1], false, bfrag, (short)0, acc[1][nt], false, false);
        }
    }
}

__global__ __launch_bounds__(THREADS)
void moe_fused_wmma(const float* __restrict__ x,
                    const unsigned int* __restrict__ wp,
                    const float* __restrict__ WgT,
                    const float* __restrict__ b1, const float* __restrict__ b2,
                    const float* __restrict__ bg,
                    float* __restrict__ out) {
    extern __shared__ char smem_raw[];
    bf16_t* Xs = (bf16_t*)smem_raw;                               // 64 x XROW bf16
    bf16_t* Hs = Xs + ROWS_WG * XROW;                             // 64 x XROW bf16
    unsigned int* Wp = (unsigned int*)(Hs + ROWS_WG * XROW);      // 3 x 512 x 20 dwords
    float* Glog = (float*)(Wp + NWPBUF * WPBUFDW);                // 64 x 8 ([row][e])
    float* Ggat = Glog + ROWS_WG * N_EXPERTS;                     // 8 x 64 ([e][row])

    const int tid  = threadIdx.x;
    const int lane = tid & 31;
    const int wave = tid >> 5;
    const int half = lane >> 4;
    const int ln   = lane & 15;
    const int mtp  = wave & 1;        // M-tile pair (rows mtp*32..+31)
    const int q8   = wave >> 1;       // 64-col slice
    const int rowbase = blockIdx.x * ROWS_WG;

    // Kick off first two weight panels (stream during staging/gating)
    async_stage_panel((unsigned long long)(size_t)wp, Wp, tid);
    async_stage_panel((unsigned long long)(size_t)wp + PANELBYTES, Wp + WPBUFDW, tid);

    // ---- Stage X tile -> LDS bf16 (padded rows) ----
    for (int it = 0; it < (ROWS_WG * D_MODEL) / THREADS; ++it) {
        int p = tid + it * THREADS;
        int r = p >> 9;
        int c = p & (D_MODEL - 1);
        Xs[(size_t)r * XROW + c] = f32_to_bf16(x[(size_t)(rowbase + r) * D_MODEL + c]);
    }

    // ---- Gating logits: vectorized f32 dot (one (row,expert) per thread) ----
    {
        int r = tid >> 3;
        int e = tid & 7;
        const float4* xr4 = (const float4*)(x + (size_t)(rowbase + r) * D_MODEL);
        const float4* wg4 = (const float4*)(WgT + (size_t)e * D_MODEL);
        float acc = bg[e];
        for (int d = 0; d < D_MODEL / 4; ++d) {
            float4 a = xr4[d], b = wg4[d];
            acc = fmaf(a.x, b.x, acc);
            acc = fmaf(a.y, b.y, acc);
            acc = fmaf(a.z, b.z, acc);
            acc = fmaf(a.w, b.w, acc);
        }
        Glog[r * N_EXPERTS + e] = acc;
    }
    __syncthreads();
    if (tid < ROWS_WG) {
        float m = Glog[tid * 8];
#pragma unroll
        for (int e = 1; e < 8; ++e) m = fmaxf(m, Glog[tid * 8 + e]);
        float ex[8], s = 0.f;
#pragma unroll
        for (int e = 0; e < 8; ++e) { ex[e] = __expf(Glog[tid * 8 + e] - m); s += ex[e]; }
        float inv = 1.f / s;
#pragma unroll
        for (int e = 0; e < 8; ++e) Ggat[e * ROWS_WG + tid] = ex[e] * inv;
    }
    // (visibility covered by barriers inside gemm_pass)

    v8f fOut[2][NTN];
#pragma unroll
    for (int mi = 0; mi < 2; ++mi)
#pragma unroll
        for (int nt = 0; nt < NTN; ++nt) fOut[mi][nt] = v8f_zero();

    int pb3 = 0;   // (global panel counter) % 3 at pass start
    for (int e = 0; e < N_EXPERTS; ++e) {
        v8f acc[2][NTN];
#pragma unroll
        for (int mi = 0; mi < 2; ++mi)
#pragma unroll
            for (int nt = 0; nt < NTN; ++nt) acc[mi][nt] = v8f_zero();

        unsigned long long w1base =
            (unsigned long long)(size_t)(wp + (size_t)e * NKSTEPS * PANELDW);
        unsigned long long w2base =
            (unsigned long long)(size_t)(wp + (size_t)(N_EXPERTS + e) * NKSTEPS * PANELDW);
        unsigned long long w1next = (e + 1 < N_EXPERTS)
            ? (unsigned long long)(size_t)(wp + (size_t)(e + 1) * NKSTEPS * PANELDW)
            : w2base;  // last pass: harmless dummy prefetch

        // ---- GEMM1: Xs @ W1[e] (tail prefetches GEMM2 panels 0-1) ----
        gemm_pass(Xs, w1base, w2base, Wp, acc, tid, mtp, q8, half, ln, pb3);
        pb3 = (pb3 + NKSTEPS) % 3;

        // ---- bias + ReLU -> Hs (bf16) ----
#pragma unroll
        for (int mi = 0; mi < 2; ++mi)
#pragma unroll
            for (int nt = 0; nt < NTN; ++nt) {
                int col = q8 * 64 + nt * 16 + ln;
                float b1v = b1[e * D_MODEL + col];
#pragma unroll
                for (int i = 0; i < 8; ++i) {
                    float h = fmaxf(acc[mi][nt][i] + b1v, 0.f);
                    Hs[(size_t)(mtp * 32 + mi * 16 + half * 8 + i) * XROW + col] =
                        f32_to_bf16(h);
                }
                acc[mi][nt] = v8f_zero();
            }

        // ---- GEMM2: Hs @ W2[e] (tail prefetches next expert's panels 0-1) ----
        gemm_pass(Hs, w2base, w1next, Wp, acc, tid, mtp, q8, half, ln, pb3);
        pb3 = (pb3 + NKSTEPS) % 3;

        // ---- gate-weighted combine (contiguous 32B gate loads) ----
#pragma unroll
        for (int mi = 0; mi < 2; ++mi) {
            const float* gp = Ggat + e * ROWS_WG + mtp * 32 + mi * 16 + half * 8;
            float g[8];
#pragma unroll
            for (int i = 0; i < 8; ++i) g[i] = gp[i];
#pragma unroll
            for (int nt = 0; nt < NTN; ++nt) {
                int col = q8 * 64 + nt * 16 + ln;
                float b2v = b2[e * D_MODEL + col];
#pragma unroll
                for (int i = 0; i < 8; ++i)
                    fOut[mi][nt][i] = fmaf(g[i], acc[mi][nt][i] + b2v, fOut[mi][nt][i]);
            }
        }
    }

    // ---- final f32 output ----
#pragma unroll
    for (int mi = 0; mi < 2; ++mi)
#pragma unroll
        for (int nt = 0; nt < NTN; ++nt) {
            int col = q8 * 64 + nt * 16 + ln;
#pragma unroll
            for (int i = 0; i < 8; ++i)
                out[(size_t)(rowbase + mtp * 32 + mi * 16 + half * 8 + i) * D_MODEL + col] =
                    fOut[mi][nt][i];
        }
}

extern "C" void kernel_launch(void* const* d_in, const int* in_sizes, int n_in,
                              void* d_out, int out_size, void* d_ws, size_t ws_size,
                              hipStream_t stream) {
    (void)n_in; (void)out_size; (void)ws_size;
    const float* x  = (const float*)d_in[0];
    const float* W1 = (const float*)d_in[1];
    const float* b1 = (const float*)d_in[2];
    const float* W2 = (const float*)d_in[3];
    const float* b2 = (const float*)d_in[4];
    const float* Wg = (const float*)d_in[5];
    const float* bg = (const float*)d_in[6];
    float* out = (float*)d_out;
    unsigned int* wp = (unsigned int*)d_ws;           // 8 MB packed weights
    float* wgt = (float*)(wp + PACKDW);               // + 16 KB transposed Wg

    int prep_total = PACK_TASKS + N_EXPERTS * D_MODEL;
    pack_weights<<<dim3((prep_total + THREADS - 1) / THREADS), dim3(THREADS), 0, stream>>>(
        W1, W2, Wg, wp, wgt);

    int rows = in_sizes[0] / D_MODEL;                 // 16384
    int nwg  = rows / ROWS_WG;                        // 256

    size_t shmem = (size_t)2 * ROWS_WG * XROW * sizeof(bf16_t)
                 + (size_t)NWPBUF * WPBUFDW * sizeof(unsigned int)
                 + (size_t)2 * ROWS_WG * N_EXPERTS * sizeof(float);

    moe_fused_wmma<<<dim3(nwg), dim3(THREADS), shmem, stream>>>(
        x, wp, wgt, b1, b2, bg, out);
}
